// ProperTGNN_90546500535093
// MI455X (gfx1250) — compile-verified, matrix operand order checked by reading, not run on hardware
//
#include <hip/hip_runtime.h>
#include <hip/hip_bf16.h>
#include <math.h>

// ---------------- problem constants ----------------
static constexpr int T_  = 4;
static constexpr int N_  = 20000;
static constexpr int E_  = 320000;
static constexpr int K_  = 64;
static constexpr int DN_ = 128;
static constexpr int DE_ = 16;
static constexpr int C_  = 64;

typedef __attribute__((ext_vector_type(16))) __bf16 v16bf;
typedef __attribute__((ext_vector_type(8)))  float  v8f;

// ---------------- small helpers ----------------
__device__ __forceinline__ void atomicMaxFloat(float* addr, float val) {
    unsigned int* ua = reinterpret_cast<unsigned int*>(addr);
    unsigned int old = *ua;
    for (;;) {
        float f = __uint_as_float(old);
        if (f >= val) break;
        unsigned int prev = atomicCAS(ua, old, __float_as_uint(val));
        if (prev == old) break;
        old = prev;
    }
}

__device__ __forceinline__ float sigf(float x) { return 1.0f / (1.0f + __expf(-x)); }

// ---------------- fill ----------------
__global__ void fill_kernel(float* __restrict__ p, float v, int n) {
    int i = blockIdx.x * blockDim.x + threadIdx.x;
    if (i < n) p[i] = v;
}

// ---------------- edge_attr mean per timestep ----------------
__global__ void ea_mean_kernel(const float* __restrict__ ea, float* __restrict__ out, int E) {
    __shared__ float s[DE_];
    const float* p = ea + (size_t)blockIdx.x * E * DE_;
    if (threadIdx.x < DE_) s[threadIdx.x] = 0.f;
    __syncthreads();
    float part[DE_];
#pragma unroll
    for (int k = 0; k < DE_; ++k) part[k] = 0.f;
    for (int e = threadIdx.x; e < E; e += blockDim.x) {
#pragma unroll
        for (int k = 0; k < DE_; ++k) part[k] += p[(size_t)e * DE_ + k];
    }
#pragma unroll
    for (int k = 0; k < DE_; ++k) atomicAdd(&s[k], part[k]);
    __syncthreads();
    if (threadIdx.x < DE_) out[blockIdx.x * DE_ + threadIdx.x] = s[threadIdx.x] / (float)E;
}

// ---------------- GEMM with bf16 WMMA: Out = X(Mx KDIM) @ W(KDIM x NCOL) + bias -----
// Compile-time KDIM/NCOL. Block covers MT*16 = 80 rows; blockDim = 32*(NCOL/16).
// Each wave owns one 16-col tile: preloads all KDIM/32 B fragments into registers
// once, then sweeps MT M-subtiles from an LDS-staged A panel -> MT*KDIM/32 WMMAs
// per wave with a single barrier.
template <int KDIM, int NCOL>
__global__ void gemm_bias_wmma(const float* __restrict__ X, const float* __restrict__ W,
                               const float* __restrict__ bias, float* __restrict__ Out) {
    constexpr int MT = 5;                       // 80 rows per block (20000 % 80 == 0)
    constexpr int KT = KDIM / 32;
    __shared__ float As[16 * MT][KDIM + 1];
    const int tid  = threadIdx.x;
    const int wave = tid >> 5;
    const int lane = tid & 31;
    const int g    = lane >> 4;                 // half-wave group
    const int ln   = lane & 15;                 // row (A) / col (B,D)
    const int row0   = blockIdx.x * (16 * MT);
    const int tile_n = wave * 16;

    // ---- preload all B fragments (register-resident weights) ----
    v16bf bfrag[KT];
#pragma unroll
    for (int kt = 0; kt < KT; ++kt) {
#pragma unroll
        for (int i = 0; i < 16; ++i) {
            int k = ((i >> 3) << 4) + (g << 3) + (i & 7);   // CDNA5 16-bit fragment K index
            bfrag[kt][i] = (__bf16)W[(size_t)(kt * 32 + k) * NCOL + tile_n + ln];
        }
    }

    // ---- stage 80 x KDIM A panel in LDS (single barrier) ----
    for (int idx = tid; idx < 16 * MT * KDIM; idx += blockDim.x) {
        int r = idx / KDIM;
        int c = idx - r * KDIM;
        As[r][c] = X[(size_t)(row0 + r) * KDIM + c];
    }
    __syncthreads();

    const float bv = bias[tile_n + ln];
#pragma unroll
    for (int mt = 0; mt < MT; ++mt) {
        v8f acc = {0.f, 0.f, 0.f, 0.f, 0.f, 0.f, 0.f, 0.f};
#pragma unroll
        for (int kt = 0; kt < KT; ++kt) {
            v16bf a;
#pragma unroll
            for (int i = 0; i < 16; ++i) {
                int k = ((i >> 3) << 4) + (g << 3) + (i & 7);
                a[i] = (__bf16)As[mt * 16 + ln][kt * 32 + k];
            }
            acc = __builtin_amdgcn_wmma_f32_16x16x32_bf16(false, a, false, bfrag[kt],
                                                          (short)0, acc, false, false);
        }
#pragma unroll
        for (int r = 0; r < 8; ++r) {
            int m = row0 + mt * 16 + r + (g << 3);
            Out[(size_t)m * NCOL + tile_n + ln] = acc[r] + bv;
        }
    }
}

// ---------------- GATv2 edge passes ----------------
// pass 1: logit per (edge, head) + segment max
__global__ void edge_logit_kernel(const float* __restrict__ xl, const float* __restrict__ xr,
                                  const int* __restrict__ ei, const float* __restrict__ ea,
                                  const float* __restrict__ ea_mean,
                                  const float* __restrict__ We, const float* __restrict__ att,
                                  float* __restrict__ logit, float* __restrict__ mx,
                                  int E, int N, int H) {
    int idx = blockIdx.x * blockDim.x + threadIdx.x;
    int total = (E + N) * H;
    if (idx >= total) return;
    int e = idx / H;
    int h = idx - e * H;
    int HC = H * C_;
    int src, dst;
    float eav[DE_];
    if (e < E) {
        src = ei[e];
        dst = ei[E + e];
#pragma unroll
        for (int k = 0; k < DE_; ++k) eav[k] = ea[(size_t)e * DE_ + k];
    } else {
        src = dst = e - E;
#pragma unroll
        for (int k = 0; k < DE_; ++k) eav[k] = ea_mean[k];
    }
    const float* xlp = xl + (size_t)src * HC + h * C_;
    const float* xrp = xr + (size_t)dst * HC + h * C_;
    const float* Wp  = We + h * C_;
    const float* ap  = att + h * C_;
    float acc = 0.f;
    for (int c = 0; c < C_; ++c) {
        float eec = 0.f;
#pragma unroll
        for (int k = 0; k < DE_; ++k) eec += eav[k] * Wp[k * HC + c];
        float m = xlp[c] + xrp[c] + eec;
        m = (m > 0.f) ? m : 0.2f * m;
        acc += m * ap[c];
    }
    logit[idx] = acc;
    atomicMaxFloat(&mx[dst * H + h], acc);
}

// pass 2: ex = exp(logit - mx[dst]); den[dst] += ex
__global__ void edge_exp_kernel(const int* __restrict__ ei, float* __restrict__ logit,
                                const float* __restrict__ mx, float* __restrict__ den,
                                int E, int N, int H) {
    int idx = blockIdx.x * blockDim.x + threadIdx.x;
    int total = (E + N) * H;
    if (idx >= total) return;
    int e = idx / H;
    int h = idx - e * H;
    int dst = (e < E) ? ei[E + e] : (e - E);
    float ex = __expf(logit[idx] - mx[dst * H + h]);
    logit[idx] = ex;
    atomicAdd(&den[dst * H + h], ex);
}

// pass 3: out[dst] += alpha * xl[src]
__global__ void edge_scatter_kernel(const float* __restrict__ xl, const int* __restrict__ ei,
                                    const float* __restrict__ logit, const float* __restrict__ den,
                                    float* __restrict__ out, int E, int N, int H) {
    int idx = blockIdx.x * blockDim.x + threadIdx.x;
    int total = (E + N) * H;
    if (idx >= total) return;
    int e = idx / H;
    int h = idx - e * H;
    int HC = H * C_;
    int src, dst;
    if (e < E) { src = ei[e]; dst = ei[E + e]; }
    else       { src = dst = e - E; }
    float alpha = logit[idx] / (den[dst * H + h] + 1e-16f);
    const float* xlp = xl + (size_t)src * HC + h * C_;
    float* op = out + (size_t)dst * HC + h * C_;
    for (int c = 0; c < C_; ++c) atomicAdd(&op[c], alpha * xlp[c]);
}

// ---------------- LayerNorm (+bias, +ReLU), in place, warp per node ----------------
__global__ void ln_relu_kernel(float* __restrict__ buf, const float* __restrict__ bias,
                               const float* __restrict__ gw, const float* __restrict__ bw,
                               int N, int D) {
    int wpb  = blockDim.x >> 5;
    int node = blockIdx.x * wpb + (threadIdx.x >> 5);
    int lane = threadIdx.x & 31;
    if (node >= N) return;
    float* p = buf + (size_t)node * D;
    int cnt = D >> 5;                       // 4 (D=128) or 2 (D=64)
    float v[4];
    float s = 0.f;
    for (int i = 0; i < cnt; ++i) {
        int j = lane + (i << 5);
        v[i] = p[j] + bias[j];
        s += v[i];
    }
    for (int o = 16; o > 0; o >>= 1) s += __shfl_xor(s, o);
    float mu = s / (float)D;
    float vs = 0.f;
    for (int i = 0; i < cnt; ++i) { float d = v[i] - mu; vs += d * d; }
    for (int o = 16; o > 0; o >>= 1) vs += __shfl_xor(vs, o);
    float rstd = rsqrtf(vs / (float)D + 1e-5f);
    for (int i = 0; i < cnt; ++i) {
        int j = lane + (i << 5);
        float y = (v[i] - mu) * rstd * gw[j] + bw[j];
        p[j] = y > 0.f ? y : 0.f;
    }
}

// ---------------- 2-layer GRU over the ue node's sequence + ue projection ----------------
__global__ void gru_kernel(const float* __restrict__ Hs, const int* __restrict__ ue_index_p, int N,
                           const float* __restrict__ Wih0, const float* __restrict__ Whh0,
                           const float* __restrict__ bih0, const float* __restrict__ bhh0,
                           const float* __restrict__ Wih1, const float* __restrict__ Whh1,
                           const float* __restrict__ bih1, const float* __restrict__ bhh1,
                           const float* __restrict__ ue_W, const float* __restrict__ ue_b,
                           float* __restrict__ u_buf) {
    __shared__ float xs[T_][C_], seqo[T_][C_], h[C_], xc[C_], gi[3 * C_], gh[3 * C_];
    int tid = threadIdx.x;
    int ue  = ue_index_p[0];
    for (int i = tid; i < T_ * C_; i += blockDim.x) {
        int t = i >> 6, c = i & 63;
        xs[t][c] = Hs[(size_t)t * N * C_ + (size_t)ue * C_ + c];
    }
    __syncthreads();
    for (int layer = 0; layer < 2; ++layer) {
        const float* Wih = layer ? Wih1 : Wih0;
        const float* Whh = layer ? Whh1 : Whh0;
        const float* bih = layer ? bih1 : bih0;
        const float* bhh = layer ? bhh1 : bhh0;
        if (tid < C_) h[tid] = 0.f;
        __syncthreads();
        for (int t = 0; t < T_; ++t) {
            if (tid < C_) xc[tid] = layer ? seqo[t][tid] : xs[t][tid];
            __syncthreads();
            if (tid < 3 * C_) {
                float a = bih[tid], b = bhh[tid];
                for (int k = 0; k < C_; ++k) {
                    a += xc[k] * Wih[tid * C_ + k];
                    b += h[k] * Whh[tid * C_ + k];
                }
                gi[tid] = a; gh[tid] = b;
            }
            __syncthreads();
            float hn = 0.f;
            if (tid < C_) {
                float r  = sigf(gi[tid] + gh[tid]);
                float z  = sigf(gi[C_ + tid] + gh[C_ + tid]);
                float ng = tanhf(gi[2 * C_ + tid] + r * gh[2 * C_ + tid]);
                hn = (1.f - z) * ng + z * h[tid];
            }
            __syncthreads();
            if (tid < C_) { h[tid] = hn; seqo[t][tid] = hn; }
            __syncthreads();
        }
    }
    if (tid < C_) {
        float u = ue_b[tid];
        for (int k = 0; k < C_; ++k) u += h[k] * ue_W[k * C_ + tid];
        u_buf[tid] = u;
    }
}

// ---------------- tower scoring head: one block per tower ----------------
__global__ void score_kernel(const float* __restrict__ Hs3, const int* __restrict__ tower_indices,
                             const float* __restrict__ u_buf,
                             const float* __restrict__ tw_W, const float* __restrict__ tw_b,
                             const float* __restrict__ s1_W, const float* __restrict__ s1_b,
                             const float* __restrict__ s2_W, const float* __restrict__ s2_b,
                             float* __restrict__ out) {
    __shared__ float trow[C_], twv[C_], us[C_], pair[3 * C_], hsc[C_];
    int c = threadIdx.x;
    int j = blockIdx.x;
    int tix = tower_indices[j];
    trow[c] = Hs3[(size_t)tix * C_ + c];
    us[c]   = u_buf[c];
    __syncthreads();
    float tw = tw_b[c];
    for (int k = 0; k < C_; ++k) tw += trow[k] * tw_W[k * C_ + c];
    twv[c] = tw;
    __syncthreads();
    for (int k = c; k < 3 * C_; k += C_)
        pair[k] = (k < C_) ? us[k] : ((k < 2 * C_) ? twv[k - C_] : us[k - 2 * C_] - twv[k - 2 * C_]);
    __syncthreads();
    float hv = s1_b[c];
    for (int k = 0; k < 3 * C_; ++k) hv += pair[k] * s1_W[k * C_ + c];
    hsc[c] = hv > 0.f ? hv : 0.f;
    __syncthreads();
    if (c == 0) {
        float s = s2_b[0];
        for (int k = 0; k < C_; ++k) s += hsc[k] * s2_W[k];
        out[j] = s;
    }
}

// ---------------- launcher ----------------
static inline void fill(float* p, float v, size_t n, hipStream_t s) {
    int blocks = (int)((n + 255) / 256);
    fill_kernel<<<blocks, 256, 0, s>>>(p, v, (int)n);
}

extern "C" void kernel_launch(void* const* d_in, const int* in_sizes, int n_in,
                              void* d_out, int out_size, void* d_ws, size_t ws_size,
                              hipStream_t stream) {
    const float* x          = (const float*)d_in[0];
    const int*   edge_index = (const int*)d_in[1];
    const float* edge_attr  = (const float*)d_in[2];
    const int*   ue_index   = (const int*)d_in[3];
    const int*   tower_idx  = (const int*)d_in[4];
    const float* g1_Wl = (const float*)d_in[5];
    const float* g1_bl = (const float*)d_in[6];
    const float* g1_Wr = (const float*)d_in[7];
    const float* g1_br = (const float*)d_in[8];
    const float* g1_We = (const float*)d_in[9];
    const float* g1_att = (const float*)d_in[10];
    const float* g1_bias = (const float*)d_in[11];
    const float* ln1_g = (const float*)d_in[12];
    const float* ln1_b = (const float*)d_in[13];
    const float* g2_Wl = (const float*)d_in[14];
    const float* g2_bl = (const float*)d_in[15];
    const float* g2_Wr = (const float*)d_in[16];
    const float* g2_br = (const float*)d_in[17];
    const float* g2_We = (const float*)d_in[18];
    const float* g2_att = (const float*)d_in[19];
    const float* g2_bias = (const float*)d_in[20];
    const float* ln2_g = (const float*)d_in[21];
    const float* ln2_b = (const float*)d_in[22];
    const float* gru_Wih0 = (const float*)d_in[23];
    const float* gru_Whh0 = (const float*)d_in[24];
    const float* gru_bih0 = (const float*)d_in[25];
    const float* gru_bhh0 = (const float*)d_in[26];
    const float* gru_Wih1 = (const float*)d_in[27];
    const float* gru_Whh1 = (const float*)d_in[28];
    const float* gru_bih1 = (const float*)d_in[29];
    const float* gru_bhh1 = (const float*)d_in[30];
    const float* ue_W = (const float*)d_in[31];
    const float* ue_b = (const float*)d_in[32];
    const float* tw_W = (const float*)d_in[33];
    const float* tw_b = (const float*)d_in[34];
    const float* s1_W = (const float*)d_in[35];
    const float* s1_b = (const float*)d_in[36];
    const float* s2_W = (const float*)d_in[37];
    const float* s2_b = (const float*)d_in[38];

    // ---- workspace carve (f32) ----
    float* ws = (float*)d_ws;
    float* ea_mean = ws;                                   // T*16
    float* xl1  = ea_mean + T_ * DE_;                      // N*128
    float* xr1  = xl1 + (size_t)N_ * 128;                  // N*128
    float* buf1 = xr1 + (size_t)N_ * 128;                  // N*128 (out1 / h1 in place)
    float* xl2  = buf1 + (size_t)N_ * 128;                 // N*64
    float* xr2  = xl2 + (size_t)N_ * C_;                   // N*64
    float* Hs   = xr2 + (size_t)N_ * C_;                   // T*N*64
    float* logitb = Hs + (size_t)T_ * N_ * C_;             // (E+N)*2
    float* mx   = logitb + (size_t)(E_ + N_) * 2;          // N*2
    float* den  = mx + (size_t)N_ * 2;                     // N*2
    float* u_buf = den + (size_t)N_ * 2;                   // 64

    ea_mean_kernel<<<T_, 256, 0, stream>>>(edge_attr, ea_mean, E_);

    const int mblk = N_ / 80;                     // 250 blocks (80 rows each, exact)
    const int eb1 = ((E_ + N_) * 2 + 255) / 256;  // layer1: H=2
    const int eb2 = ((E_ + N_) * 1 + 255) / 256;  // layer2: H=1
    const int lnb = (N_ + 7) / 8;                 // warp-per-node, 8 nodes/block

    for (int t = 0; t < T_; ++t) {
        const float* x_t  = x + (size_t)t * N_ * DN_;
        const int*   ei_t = edge_index + (size_t)t * 2 * E_;
        const float* ea_t = edge_attr + (size_t)t * E_ * DE_;
        const float* eam_t = ea_mean + t * DE_;
        float* Hs_t = Hs + (size_t)t * N_ * C_;

        // ---- layer 1 (H=2, HC=128) ----
        gemm_bias_wmma<128, 128><<<mblk, 256, 0, stream>>>(x_t, g1_Wl, g1_bl, xl1);
        gemm_bias_wmma<128, 128><<<mblk, 256, 0, stream>>>(x_t, g1_Wr, g1_br, xr1);
        fill(mx, -1e30f, (size_t)N_ * 2, stream);
        fill(den, 0.f, (size_t)N_ * 2, stream);
        fill(buf1, 0.f, (size_t)N_ * 128, stream);
        edge_logit_kernel<<<eb1, 256, 0, stream>>>(xl1, xr1, ei_t, ea_t, eam_t,
                                                   g1_We, g1_att, logitb, mx, E_, N_, 2);
        edge_exp_kernel<<<eb1, 256, 0, stream>>>(ei_t, logitb, mx, den, E_, N_, 2);
        edge_scatter_kernel<<<eb1, 256, 0, stream>>>(xl1, ei_t, logitb, den, buf1, E_, N_, 2);
        ln_relu_kernel<<<lnb, 256, 0, stream>>>(buf1, g1_bias, ln1_g, ln1_b, N_, 128);

        // ---- layer 2 (H=1, HC=64) ----
        gemm_bias_wmma<128, 64><<<mblk, 128, 0, stream>>>(buf1, g2_Wl, g2_bl, xl2);
        gemm_bias_wmma<128, 64><<<mblk, 128, 0, stream>>>(buf1, g2_Wr, g2_br, xr2);
        fill(mx, -1e30f, (size_t)N_, stream);
        fill(den, 0.f, (size_t)N_, stream);
        fill(Hs_t, 0.f, (size_t)N_ * C_, stream);
        edge_logit_kernel<<<eb2, 256, 0, stream>>>(xl2, xr2, ei_t, ea_t, eam_t,
                                                   g2_We, g2_att, logitb, mx, E_, N_, 1);
        edge_exp_kernel<<<eb2, 256, 0, stream>>>(ei_t, logitb, mx, den, E_, N_, 1);
        edge_scatter_kernel<<<eb2, 256, 0, stream>>>(xl2, ei_t, logitb, den, Hs_t, E_, N_, 1);
        ln_relu_kernel<<<lnb, 256, 0, stream>>>(Hs_t, g2_bias, ln2_g, ln2_b, N_, 64);
    }

    gru_kernel<<<1, 192, 0, stream>>>(Hs, ue_index, N_,
                                      gru_Wih0, gru_Whh0, gru_bih0, gru_bhh0,
                                      gru_Wih1, gru_Whh1, gru_bih1, gru_bhh1,
                                      ue_W, ue_b, u_buf);

    const float* Hs3 = Hs + (size_t)(T_ - 1) * N_ * C_;
    score_kernel<<<K_, C_, 0, stream>>>(Hs3, tower_idx, u_buf, tw_W, tw_b,
                                        s1_W, s1_b, s2_W, s2_b, (float*)d_out);
}
